// LSTMfs__46308337386349
// MI455X (gfx1250) — matrix operand order repaired and checked by reading, not compiled, not run on hardware
//
#include <hip/hip_runtime.h>
#include <hip/hip_bf16.h>

typedef __bf16 bf16_t;
typedef bf16_t v16bf __attribute__((ext_vector_type(16)));
typedef bf16_t v8bf  __attribute__((ext_vector_type(8)));
typedef float  v8f   __attribute__((ext_vector_type(8)));

#define HID   256
#define BATCH 32
#define SEQ   1024
#define G4    1024   // 4*HID

// ---------------- fast transcendental helpers ----------------
#if __has_builtin(__builtin_amdgcn_tanhf)
__device__ __forceinline__ float fast_tanh(float x) { return __builtin_amdgcn_tanhf(x); }
#elif __has_builtin(__builtin_amdgcn_tanh_f32)
__device__ __forceinline__ float fast_tanh(float x) { return __builtin_amdgcn_tanh_f32(x); }
#else
__device__ __forceinline__ float fast_tanh(float x) { return tanhf(x); }
#endif
// exact identity: sigmoid(x) = 0.5*tanh(x/2) + 0.5
__device__ __forceinline__ float sigf(float x) { return fmaf(0.5f, fast_tanh(0.5f * x), 0.5f); }

// ---- WMMA fragment loaders (layouts per CDNA5 ISA §7.12.2) ----

// A-matrix 16x32 bf16 from a row-major row: lane m (and m+16) hold row m;
// K = koff+[0..7] and koff+[16..23], koff = (lane>=16)*8.
__device__ __forceinline__ v16bf load_a_frag(const bf16_t* rowbase, int lane) {
    const int koff = (lane >> 4) * 8;
    v8bf lo = *(const v8bf*)(rowbase + koff);
    v8bf hi = *(const v8bf*)(rowbase + koff + 16);
    v16bf a;
#pragma unroll
    for (int i = 0; i < 8; ++i) { a[i] = lo[i]; a[i + 8] = hi[i]; }
    return a;
}

__device__ __forceinline__ v8f wmma_bf16(v16bf a, v16bf b, v8f c) {
    return __builtin_amdgcn_wmma_f32_16x16x32_bf16(false, a, false, b, (short)0, c,
                                                   false, false);
}

// ---------------- Kernel 1: path table -> bf16 ----------------
__global__ __launch_bounds__(HID) void path_kernel(
    const int* __restrict__ anc, const float* __restrict__ weight,
    const float* __restrict__ emb, int N, int D, bf16_t* __restrict__ path_bf)
{
    const int n = blockIdx.x;
    const int h = threadIdx.x;
    float acc = 0.0f, cw = 1.0f;
    for (int d = 0; d < D; ++d) {
        const int a = anc[(size_t)n * D + d];
        if (a < 0) break;
        acc += cw * emb[(size_t)a * HID + h];
        cw *= weight[a];
    }
    path_bf[(size_t)n * HID + h] = (bf16_t)acc;
}

// ------- Kernel 2: weight fp32 -> bf16, swizzled to B-fragment-major -------
// dst[((gt*8 + kc)*32 + lane)*16 + i] = (bf16) src[g*256 + k]  with
// g = gt*16 + (lane&15), k = kc*32 + (lane>>4)*8 + i + (i>=8)*8.
// A wave's B-frag load becomes one fully-coalesced contiguous 1KB request.
__global__ __launch_bounds__(256) void wswz_kernel(const float* __restrict__ src,
                                                   bf16_t* __restrict__ dst)
{
    const int idx  = blockIdx.x * blockDim.x + threadIdx.x;  // < 64*8*32
    const int lane = idx & 31;
    const int kc   = (idx >> 5) & 7;
    const int gt   = idx >> 8;
    const int g    = gt * 16 + (lane & 15);
    const int k0   = kc * 32 + (lane >> 4) * 8;
    v16bf tmp;
#pragma unroll
    for (int i = 0; i < 16; ++i) tmp[i] = (bf16_t)src[(size_t)g * HID + k0 + i + (i >= 8 ? 8 : 0)];
    *(v16bf*)(dst + (size_t)idx * 16) = tmp;
}

// ------- Kernel 3: x_proj = gather(path) @ W_ih^T + b  (WMMA, fragment-major I/O) -------
// Rows r = t*BATCH + b (emb_seq is [T,B,H]); M = 32768, K = 256, N = 1024.
// Output layout: [t][gtile(64)][mt(2)][lane(32)][v(8)] floats.
__global__ __launch_bounds__(256) void xproj_kernel(
    const int* __restrict__ events, const bf16_t* __restrict__ path,
    const bf16_t* __restrict__ WihF, const float* __restrict__ b_ih,
    const float* __restrict__ b_hh, float* __restrict__ xproj)
{
    const int lane  = threadIdx.x & 31;
    const int wave  = threadIdx.x >> 5;
    const int gtile = blockIdx.x * 8 + wave;   // 0..63
    const int mtile = blockIdx.y;              // 0..2047  (t = mtile>>1, mt = mtile&1)

    const int mrow = mtile * 16 + (lane & 15); // global A row
    const int t = mrow >> 5, b = mrow & 31;
    const int ev = events[b * SEQ + t];
    const bf16_t* arow  = path + (size_t)ev * HID;
    const bf16_t* bfrag = WihF + ((size_t)gtile * 8 * 32 + lane) * 16;

    v8f acc = {};
#pragma unroll
    for (int kc = 0; kc < 8; ++kc) {
        v16bf a  = load_a_frag(arow + kc * 32, lane);
        v16bf bb = *(const v16bf*)(bfrag + (size_t)kc * 32 * 16);
        acc = wmma_bf16(a, bb, acc);
    }
    const int g = gtile * 16 + (lane & 15);
    const float bias = b_ih[g] + b_hh[g];
#pragma unroll
    for (int v = 0; v < 8; ++v) acc[v] += bias;

    const size_t frag = ((size_t)(mtile >> 1) * 64 + gtile) * 2 + (mtile & 1);
    *(v8f*)(xproj + frag * 256 + lane * 8) = acc;
}

// ---------------- Kernel 4: persistent LSTM scan (single WGP) ----------------
// 16 waves; wave w owns hidden columns n in [16w, 16w+16) across ALL FOUR gates:
// g-tiles {w, 16+w, 32+w, 48+w} x 2 m-tiles x 8 K-chunks = 64 WMMAs/step/wave.
// Cell state c is register-resident; h is double-buffered in LDS in A-FRAGMENT-MAJOR
// layout so each A-frag is one 32B ds_load.  W_hh for gates i,f (256 KB) is cached
// in LDS once; gates g,o stream from L2 -> per-step L2 read volume halved and
// fragment feeds split across the independent DS and VMEM paths.
// LDS: 256 KB W_hh + 2 x 16 KB h = 288 KB (needs CDNA5's 320 KB WGP LDS).
__global__ __launch_bounds__(512) void lstm_scan_kernel(
    const float* __restrict__ xprojf, const bf16_t* __restrict__ WhhF,
    float* __restrict__ out)
{
    __shared__ __attribute__((aligned(64))) bf16_t whh_lds[32 * 8 * 32 * 16]; // 256 KB
    __shared__ __attribute__((aligned(64))) bf16_t h_buf[2][2 * 8 * 32 * 16]; // 2x16 KB

    const int tid  = threadIdx.x;
    const int lane = tid & 31;
    const int w    = tid >> 5;        // 0..15
    const int l15  = lane & 15;
    const int lhi  = lane >> 4;       // 0/1

    // one-time: cache gate-i/f half of fragment-major W_hh in LDS (8192 x 32B)
    for (int i = tid; i < 32 * 8 * 32; i += 512)
        *(v16bf*)(whh_lds + (size_t)i * 16) = *(const v16bf*)(WhhF + (size_t)i * 16);
    for (int i = tid; i < 2 * 8 * 32 * 16; i += 512) h_buf[0][i] = (bf16_t)0.0f;
    __syncthreads();

    const int n = w * 16 + l15;       // owned hidden column

    // --- precompute static h-producer LDS offsets (A-fragment-major target) ---
    const int kc_own = w >> 1;
    const int kk     = (w & 1) * 16 + l15;
    int lhi_p, i_p;
    if (kk < 8)       { lhi_p = 0; i_p = kk;      }
    else if (kk < 16) { lhi_p = 1; i_p = kk - 8;  }
    else if (kk < 24) { lhi_p = 0; i_p = kk - 8;  }
    else              { lhi_p = 1; i_p = kk - 16; }
    const int wbase0 = (((0 * 8 + kc_own) * 32 + lhi_p * 16 + lhi * 8) * 16) + i_p;
    const int wbase1 = (((1 * 8 + kc_own) * 32 + lhi_p * 16 + lhi * 8) * 16) + i_p;

    v8f c0 = {}, c1 = {};             // register-resident cell state

    for (int t = 0; t < SEQ; ++t) {
        const float* xp = xprojf + (size_t)t * (BATCH * G4);
        if (t + 1 < SEQ)  // next step's 128 KB fragment block: 512 threads x 256 B
            __builtin_prefetch(xp + BATCH * G4 + tid * 64, 0, 1);

        const bf16_t* hrd = h_buf[t & 1];
        bf16_t*       hwr = h_buf[(t + 1) & 1];

        // --- init accumulators from fragment-major x_proj (single-use: NT loads) ---
        v8f acc[4][2];
#pragma unroll
        for (int g = 0; g < 4; ++g) {
            const int gt = g * 16 + w;
#pragma unroll
            for (int mt = 0; mt < 2; ++mt)
                acc[g][mt] = __builtin_nontemporal_load(
                    (const v8f*)(xp + (size_t)(gt * 2 + mt) * 256 + lane * 8));
        }

        // --- gates += h @ W_hh^T : K = 256 in 8 chunks of 32 ---
#pragma unroll
        for (int kc = 0; kc < 8; ++kc) {
            v16bf a0 = *(const v16bf*)(&hrd[((0 * 8 + kc) * 32 + lane) * 16]);
            v16bf a1 = *(const v16bf*)(&hrd[((1 * 8 + kc) * 32 + lane) * 16]);
#pragma unroll
            for (int g = 0; g < 2; ++g) {  // gates i,f: B frags from LDS
                const int gt = g * 16 + w;
                v16bf bb = *(const v16bf*)(whh_lds + (size_t)(((gt * 8 + kc) * 32) + lane) * 16);
                acc[g][0] = wmma_bf16(a0, bb, acc[g][0]);
                acc[g][1] = wmma_bf16(a1, bb, acc[g][1]);
            }
#pragma unroll
            for (int g = 2; g < 4; ++g) {  // gates g,o: B frags stream from L2
                const int gt = g * 16 + w;
                v16bf bb = *(const v16bf*)(WhhF + (size_t)(((gt * 8 + kc) * 32) + lane) * 16);
                acc[g][0] = wmma_bf16(a0, bb, acc[g][0]);
                acc[g][1] = wmma_bf16(a1, bb, acc[g][1]);
            }
        }

        // --- LSTM cell update entirely in registers (i,f,g,o all in this lane) ---
        float* outp = out + (size_t)t * (BATCH * HID);
#pragma unroll
        for (int mt = 0; mt < 2; ++mt) {
            v8f* cp = mt ? &c1 : &c0;
            const int wbase = mt ? wbase1 : wbase0;
#pragma unroll
            for (int v = 0; v < 8; ++v) {
                const int b  = mt * 16 + lhi * 8 + v;      // batch row (C layout)
                const float gi = acc[0][mt][v];
                const float gf = acc[1][mt][v];
                const float gg = acc[2][mt][v];
                const float go = acc[3][mt][v];
                const float c  = sigf(gf) * (*cp)[v] + sigf(gi) * fast_tanh(gg);
                const float h  = sigf(go) * fast_tanh(c);
                (*cp)[v] = c;
                hwr[wbase + v * 16] = (bf16_t)h;           // A-fragment-major h store
                __builtin_nontemporal_store(h, &outp[b * HID + n]);
            }
        }
        __syncthreads();   // h writes visible before next step's A-frag reads
    }
}

extern "C" void kernel_launch(void* const* d_in, const int* in_sizes, int n_in,
                              void* d_out, int out_size, void* d_ws, size_t ws_size,
                              hipStream_t stream) {
    const int*   events = (const int*)d_in[0];
    const int*   anc    = (const int*)d_in[1];
    const float* weight = (const float*)d_in[2];
    const float* emb    = (const float*)d_in[3];
    const float* W_ih   = (const float*)d_in[4];
    const float* W_hh   = (const float*)d_in[5];
    const float* b_ih   = (const float*)d_in[6];
    const float* b_hh   = (const float*)d_in[7];

    const int N = in_sizes[2];          // N_TYPES
    const int D = in_sizes[1] / N;      // ancestor depth

    char* ws = (char*)d_ws;
    bf16_t* path_bf = (bf16_t*)ws;                       // N*256 bf16
    bf16_t* WihF    = path_bf + (size_t)N * HID;         // 64*8*32*16 bf16 (frag-major)
    bf16_t* WhhF    = WihF + (size_t)G4 * HID;           // 64*8*32*16 bf16 (frag-major)
    size_t off = (size_t)((char*)(WhhF + (size_t)G4 * HID) - ws);
    off = (off + 255) & ~(size_t)255;
    float* xproj = (float*)(ws + off);                   // 32768*1024 f32 (fragment-major)

    path_kernel<<<N, HID, 0, stream>>>(anc, weight, emb, N, D, path_bf);

    wswz_kernel<<<64, 256, 0, stream>>>(W_ih, WihF);     // 64*8*32 = 16384 threads
    wswz_kernel<<<64, 256, 0, stream>>>(W_hh, WhhF);

    xproj_kernel<<<dim3(8, (SEQ * BATCH) / 16), 256, 0, stream>>>(
        events, path_bf, WihF, b_ih, b_hh, xproj);

    lstm_scan_kernel<<<1, 512, 0, stream>>>(xproj, WhhF, (float*)d_out);
}
